// PerceiverAttention_24378234372283
// MI455X (gfx1250) — matrix operand order repaired and checked by reading, not compiled, not run on hardware
//
#include <hip/hip_runtime.h>
#include <math.h>

// ---------------------------------------------------------------------------
// PerceiverAttention for MI455X (gfx1250, wave32, WMMA).
// f16 operands + f32 accumulation via v_wmma_f32_16x16x32_f16.
// ---------------------------------------------------------------------------

typedef __attribute__((ext_vector_type(16))) _Float16 v16h;
typedef __attribute__((ext_vector_type(8)))  float    v8f;
typedef __attribute__((ext_vector_type(4)))  unsigned int u32x4;

union F16Frag { v16h v; u32x4 q[2]; };

struct alignas(8) H4 { _Float16 a, b, c, d; };

#define WMMA_F16(A, B, C) \
  __builtin_amdgcn_wmma_f32_16x16x32_f16(false, (A), false, (B), (short)0, (C), false, false)

// A-fragment (16x32 f16, M x K). Memory: row-major, row stride = ld halves.
// lane<16 : row=lane,     K = 0..15
// lane>=16: row=lane-16,  K = 16..31
__device__ __forceinline__ v16h load_a_frag(const _Float16* __restrict__ base,
                                            int ld, int row, int k0, int lane) {
  const bool lo = lane < 16;
  const _Float16* p = base + (size_t)row * ld + k0;
  F16Frag f;
  f.q[0] = *(const u32x4*)(p + (lo ? 0 : 16));
  f.q[1] = *(const u32x4*)(p + (lo ? 8 : 24));
  return f.v;
}

// B-fragment (32x16 f16, K x N) from a K-contiguous ("transposed") layout:
// column `col` lives at bt + col*ld, contiguous along K.
// lane<16 : col=lane,     K = 0..15
// lane>=16: col=lane-16,  K = 16..31
__device__ __forceinline__ v16h load_b_frag(const _Float16* __restrict__ bt,
                                            int ld, int col, int k0, int lane) {
  const bool lo = lane < 16;
  const _Float16* p = bt + (size_t)col * ld + k0 + (lo ? 0 : 16);
  F16Frag f;
  f.q[0] = *(const u32x4*)(p);
  f.q[1] = *(const u32x4*)(p + 8);
  return f.v;
}

// ---------------------------------------------------------------------------
// Weight prep: fp32 (K,N) row-major  ->  f16 (N,K) row-major (transposed)
// ---------------------------------------------------------------------------
__global__ __launch_bounds__(256) void convert_transpose_kernel(
    const float* __restrict__ in, _Float16* __restrict__ out, int K, int N) {
  int t = blockIdx.x * 256 + threadIdx.x;
  if (t >= N * K) return;
  int n = t / K;
  int k = t - n * K;
  out[t] = (_Float16)in[(size_t)k * N + n];
}

// ---------------------------------------------------------------------------
// LayerNorm rows of width 1024, fp32 in -> f16 out. One block (256 thr) per row.
// ---------------------------------------------------------------------------
__global__ __launch_bounds__(256) void layernorm_kernel(
    const float* __restrict__ x, const float* __restrict__ w,
    const float* __restrict__ bias, _Float16* __restrict__ y) {
  const int row = blockIdx.x;
  const int t = threadIdx.x;
  const float4 v = ((const float4*)(x + (size_t)row * 1024))[t];

  float sum = v.x + v.y + v.z + v.w;
  float sq  = v.x * v.x + v.y * v.y + v.z * v.z + v.w * v.w;
#pragma unroll
  for (int off = 16; off >= 1; off >>= 1) {
    sum += __shfl_xor(sum, off, 32);
    sq  += __shfl_xor(sq,  off, 32);
  }
  __shared__ float sSum[8], sSq[8];
  const int wave = t >> 5, lane = t & 31;
  if (lane == 0) { sSum[wave] = sum; sSq[wave] = sq; }
  __syncthreads();
  float tot = 0.f, totq = 0.f;
#pragma unroll
  for (int i = 0; i < 8; ++i) { tot += sSum[i]; totq += sSq[i]; }
  const float mean = tot * (1.0f / 1024.0f);
  const float var  = totq * (1.0f / 1024.0f) - mean * mean;
  const float rstd = rsqrtf(var + 1e-5f);

  const float4 wv = ((const float4*)w)[t];
  const float4 bv = ((const float4*)bias)[t];
  H4 o;
  o.a = (_Float16)((v.x - mean) * rstd * wv.x + bv.x);
  o.b = (_Float16)((v.y - mean) * rstd * wv.y + bv.y);
  o.c = (_Float16)((v.z - mean) * rstd * wv.z + bv.z);
  o.d = (_Float16)((v.w - mean) * rstd * wv.w + bv.w);
  ((H4*)(y + (size_t)row * 1024))[t] = o;
}

// ---------------------------------------------------------------------------
// Generic WMMA GEMM: C(M,N) = A(M,K) @ B(K,N), A f16 row-major, B given as
// Bt (N,K) f16 row-major. 8 waves/block; each wave computes a 32(M) x 64(N)
// tile (2 A-frags reuse each B-frag -> 8 WMMAs per 12 b128 loads per K-step);
// block tiles 256(M) x 64(N).
// MODE 0: q epilogue  (AdaLN modulation + attn-scale, f16 out, pitch N)
// MODE 1: kv epilogue (split k / v; v stored transposed per head (b,h,d,n1))
// MODE 2: fp32 output (final projection)
// ---------------------------------------------------------------------------
template <int MODE>
__global__ __launch_bounds__(256) void wmma_gemm_kernel(
    const _Float16* __restrict__ A, const _Float16* __restrict__ Bt,
    int M, int N, int K,
    const float* __restrict__ modScale, const float* __restrict__ modShift,
    void* __restrict__ out0, void* __restrict__ out1) {
  const int lane  = threadIdx.x & 31;
  const int wave  = threadIdx.x >> 5;
  const int mBase = blockIdx.y * 256 + wave * 32;
  const int nBase = blockIdx.x * 64;
  const bool lo   = lane < 16;
  const int ln16  = lane & 15;
  const int row0  = mBase + ln16;
  const int row1  = mBase + 16 + ln16;

  v8f acc[2][4] = {};
  for (int k0 = 0; k0 < K; k0 += 32) {
    // prefetch next K-step (speculative: faults silently dropped)
    __builtin_prefetch(A + (size_t)row0 * K + k0 + 32, 0, 3);
    __builtin_prefetch(A + (size_t)row1 * K + k0 + 32, 0, 3);
    __builtin_prefetch(Bt + (size_t)(nBase + lane) * K + k0 + 32, 0, 3);
    __builtin_prefetch(Bt + (size_t)(nBase + 32 + lane) * K + k0 + 32, 0, 3);

    const v16h a0 = load_a_frag(A, K, row0, k0, lane);
    const v16h a1 = load_a_frag(A, K, row1, k0, lane);
#pragma unroll
    for (int g = 0; g < 4; ++g) {
      const v16h b = load_b_frag(Bt, K, nBase + g * 16 + ln16, k0, lane);
      acc[0][g] = WMMA_F16(a0, b, acc[0][g]);
      acc[1][g] = WMMA_F16(a1, b, acc[1][g]);
    }
  }

#pragma unroll
  for (int i = 0; i < 2; ++i) {
#pragma unroll
    for (int g = 0; g < 4; ++g) {
#pragma unroll
      for (int r = 0; r < 8; ++r) {
        const int rr = mBase + i * 16 + r + (lo ? 0 : 8);
        const int cc = nBase + g * 16 + ln16;
        float v = acc[i][g][r];
        if constexpr (MODE == 0) {
          const int bb = rr >> 8;   // row / 256  (batch)
          const int hh = cc >> 6;   // col / 64   (head)
          v = v * (1.0f + modScale[bb * 16 + hh]) + modShift[bb * 16 + hh];
          v *= 0.125f;              // DIM_HEAD^-0.5
          ((_Float16*)out0)[(size_t)rr * N + cc] = (_Float16)v;
        } else if constexpr (MODE == 1) {
          const int bb = rr >> 12;  // row / 4096 (batch)
          const int jj = rr & 4095; // token within batch
          if (cc < 1024) {          // K half: row-major (b*n1, INNER)
            ((_Float16*)out0)[(size_t)rr * 1024 + cc] = (_Float16)v;
          } else {                  // V half: transposed (b,h,d,n1)
            const int c2 = cc - 1024;
            const int hh = c2 >> 6;
            const int dd = c2 & 63;
            ((_Float16*)out1)[(((size_t)(bb * 16 + hh) * 64 + dd) << 12) + jj] =
                (_Float16)v;
          }
        } else {
          ((float*)out0)[(size_t)rr * N + cc] = v;
        }
      }
    }
  }
}

// ---------------------------------------------------------------------------
// Flash attention: one wave per (b, h, 16-query tile). Streams 4096 keys in
// chunks of 32 with online softmax; P re-packed through LDS into an A-frag.
// q_h: (b*256, 1024) f16, pre-modulated & pre-scaled.
// k_h: (b*4096, 1024) f16.   v_t: (b,h,64,4096) f16.
// attn_h out: (b*256, 1024) f16.
// ---------------------------------------------------------------------------
__global__ __launch_bounds__(32) void flash_attn_kernel(
    const _Float16* __restrict__ q_h, const _Float16* __restrict__ k_h,
    const _Float16* __restrict__ v_t, _Float16* __restrict__ attn_h) {
  __shared__ _Float16 Pt[16 * 32];  // P tile: rows = q rows, cols = 32 keys

  const int lane = threadIdx.x;
  const int qt = blockIdx.x & 15;
  const int h  = (blockIdx.x >> 4) & 15;
  const int b  = blockIdx.x >> 8;
  const bool lo  = lane < 16;
  const int ln16 = lane & 15;

  const int qrow = b * 256 + qt * 16 + ln16;
  const v16h aq0 = load_a_frag(q_h, 1024, qrow, h * 64 + 0,  lane);
  const v16h aq1 = load_a_frag(q_h, 1024, qrow, h * 64 + 32, lane);

  const _Float16* kb = k_h + (size_t)b * 4096 * 1024;
  const _Float16* vb = v_t + ((size_t)(b * 16 + h) * 64) * 4096;

  float m[8], l[8];
  v8f o[4] = {};
#pragma unroll
  for (int r = 0; r < 8; ++r) { m[r] = -INFINITY; l[r] = 0.0f; }

  const v8f zeroc = {};
  for (int j0 = 0; j0 < 4096; j0 += 32) {
    // prefetch next chunk's keys (32 rows x first/second 32 dims) and values
    __builtin_prefetch(kb + (size_t)(j0 + 32 + lane) * 1024 + h * 64,      0, 3);
    __builtin_prefetch(kb + (size_t)(j0 + 32 + lane) * 1024 + h * 64 + 32, 0, 3);
    __builtin_prefetch(vb + (size_t)lane        * 4096 + j0 + 32, 0, 3);
    __builtin_prefetch(vb + (size_t)(32 + lane) * 4096 + j0 + 32, 0, 3);

    // ---- scores: two 16x16 tiles (keys j0..+15, j0+16..+31), K=64 over dh
    v8f c0 = zeroc, c1 = zeroc;
    c0 = WMMA_F16(aq0, load_b_frag(kb, 1024, j0 + ln16,      h * 64 + 0,  lane), c0);
    c0 = WMMA_F16(aq1, load_b_frag(kb, 1024, j0 + ln16,      h * 64 + 32, lane), c0);
    c1 = WMMA_F16(aq0, load_b_frag(kb, 1024, j0 + 16 + ln16, h * 64 + 0,  lane), c1);
    c1 = WMMA_F16(aq1, load_b_frag(kb, 1024, j0 + 16 + ln16, h * 64 + 32, lane), c1);

    // ---- online softmax; each C element r lives in row (r + 8*hi), col ln16
#pragma unroll
    for (int r = 0; r < 8; ++r) {
      const float s0 = c0[r], s1 = c1[r];
      float tmax = fmaxf(s0, s1);
#pragma unroll
      for (int off = 8; off >= 1; off >>= 1)
        tmax = fmaxf(tmax, __shfl_xor(tmax, off, 32));  // stays in 16-lane half
      const float mnew  = fmaxf(m[r], tmax);
      const float alpha = __expf(m[r] - mnew);
      const float p0 = __expf(s0 - mnew);
      const float p1 = __expf(s1 - mnew);
      float rs = p0 + p1;
#pragma unroll
      for (int off = 8; off >= 1; off >>= 1) rs += __shfl_xor(rs, off, 32);
      l[r] = l[r] * alpha + rs;
      m[r] = mnew;
#pragma unroll
      for (int g = 0; g < 4; ++g) o[g][r] *= alpha;  // rescale O rows

      const int rrow = r + (lo ? 0 : 8);
      Pt[rrow * 32 + ln16]      = (_Float16)p0;
      Pt[rrow * 32 + 16 + ln16] = (_Float16)p1;
    }
    __syncthreads();

    // ---- O += P (16x32) @ V-chunk (32x64)
    const v16h ap = load_a_frag((const _Float16*)Pt, 32, ln16, 0, lane);
#pragma unroll
    for (int g = 0; g < 4; ++g) {
      const v16h bv = load_b_frag(vb, 4096, g * 16 + ln16, j0, lane);
      o[g] = WMMA_F16(ap, bv, o[g]);
    }
    __syncthreads();
  }

  // ---- normalize and emit (b*256 + i, h*64 + d) f16
#pragma unroll
  for (int g = 0; g < 4; ++g) {
#pragma unroll
    for (int r = 0; r < 8; ++r) {
      const int i  = qt * 16 + r + (lo ? 0 : 8);
      const int cc = h * 64 + g * 16 + ln16;
      attn_h[(size_t)(b * 256 + i) * 1024 + cc] = (_Float16)(o[g][r] / l[r]);
    }
  }
}

// ---------------------------------------------------------------------------
// Host-side orchestration
// ---------------------------------------------------------------------------
extern "C" void kernel_launch(void* const* d_in, const int* in_sizes, int n_in,
                              void* d_out, int out_size, void* d_ws, size_t ws_size,
                              hipStream_t stream) {
  (void)in_sizes; (void)n_in; (void)out_size; (void)ws_size;

  const float* x       = (const float*)d_in[0];   // (4,4096,1024)
  const float* latents = (const float*)d_in[1];   // (4,256,1024)
  const float* shift   = (const float*)d_in[2];   // (4,16)
  const float* scale   = (const float*)d_in[3];   // (4,16)
  const float* ln1_w   = (const float*)d_in[4];
  const float* ln1_b   = (const float*)d_in[5];
  const float* ln2_w   = (const float*)d_in[6];
  const float* ln2_b   = (const float*)d_in[7];
  const float* Wq      = (const float*)d_in[8];   // (1024,1024)
  const float* Wkv     = (const float*)d_in[9];   // (1024,2048)
  const float* Wout    = (const float*)d_in[10];  // (1024,1024)

  char* ws = (char*)d_ws;
  _Float16* xn_h   = (_Float16*)ws; ws += (size_t)16384 * 1024 * 2;  // 32 MB
  _Float16* ln_h   = (_Float16*)ws; ws += (size_t)1024  * 1024 * 2;  //  2 MB
  _Float16* q_h    = (_Float16*)ws; ws += (size_t)1024  * 1024 * 2;  //  2 MB
  _Float16* k_h    = (_Float16*)ws; ws += (size_t)16384 * 1024 * 2;  // 32 MB
  _Float16* v_t    = (_Float16*)ws; ws += (size_t)16384 * 1024 * 2;  // 32 MB
  _Float16* attn_h = (_Float16*)ws; ws += (size_t)1024  * 1024 * 2;  //  2 MB
  _Float16* Wq_t   = (_Float16*)ws; ws += (size_t)1024  * 1024 * 2;  //  2 MB
  _Float16* Wkv_t  = (_Float16*)ws; ws += (size_t)2048  * 1024 * 2;  //  4 MB
  _Float16* Wout_t = (_Float16*)ws; ws += (size_t)1024  * 1024 * 2;  //  2 MB

  // 1) weight convert+transpose to f16 (N,K)
  convert_transpose_kernel<<<(1024 * 1024 + 255) / 256, 256, 0, stream>>>(Wq,   Wq_t,   1024, 1024);
  convert_transpose_kernel<<<(2048 * 1024 + 255) / 256, 256, 0, stream>>>(Wkv,  Wkv_t,  1024, 2048);
  convert_transpose_kernel<<<(1024 * 1024 + 255) / 256, 256, 0, stream>>>(Wout, Wout_t, 1024, 1024);

  // 2) layernorms -> f16 activations
  layernorm_kernel<<<16384, 256, 0, stream>>>(x,       ln1_w, ln1_b, xn_h);
  layernorm_kernel<<<1024,  256, 0, stream>>>(latents, ln2_w, ln2_b, ln_h);

  // 3) q = ln @ Wq, fused AdaLN modulation + 1/sqrt(dh)
  wmma_gemm_kernel<0><<<dim3(16, 4), 256, 0, stream>>>(
      ln_h, Wq_t, 1024, 1024, 1024, scale, shift, q_h, nullptr);

  // 4) kv = xn @ Wkv, split into k (row-major) and v (head-transposed)
  wmma_gemm_kernel<1><<<dim3(32, 64), 256, 0, stream>>>(
      xn_h, Wkv_t, 16384, 2048, 1024, nullptr, nullptr, k_h, v_t);

  // 5) flash attention: b*h*(n2/16) = 1024 waves
  flash_attn_kernel<<<1024, 32, 0, stream>>>(q_h, k_h, v_t, attn_h);

  // 6) out = attn @ Wout -> fp32 d_out
  wmma_gemm_kernel<2><<<dim3(16, 4), 256, 0, stream>>>(
      attn_h, Wout_t, 1024, 1024, 1024, nullptr, nullptr, d_out, nullptr);
}
